// FilterDetections_65189013618875
// MI455X (gfx1250) — compile-verified, matrix operand order checked by reading, not run on hardware
//
#include <hip/hip_runtime.h>
#include <hip/hip_bf16.h>

// ---------------------------------------------------------------------------
// FilterDetections for MI455X (gfx1250, wave32).
//
// B=2, N=2048, C=8.  ~4MB of inputs -> fully L2-resident (192MB); the kernel
// is latency-bound.  gfx1250-specific paths used:
//   * v_wmma_f32_16x16x32_f16 for the row-count reduction row_cnt = ind2@ones
//     (indicator {0,1} exact in f16, counts <= 2048 exact in f32 acc).
//   * Tensor Data Mover (tensor_load_to_lds + s_wait_tensorcnt) to DMA the
//     per-pair colfirst[] array into LDS while the wave stages conf[] itself.
//   * ds_load_b128 LDS staging for the O(N^2) IoU scan and WMMA fragments.
//   * global_prefetch_b8 via __builtin_prefetch.
// ---------------------------------------------------------------------------

#define BATCH   2
#define NBOX    2048
#define NCLS    8
#define NPAIR   (BATCH * NCLS)        // 16
#define PN      (NPAIR * NBOX)        // 32768
#define CN      (NCLS * NBOX)         // 16384 (per-batch flattened scores)
#define MAXDET  300
#define SCORE_TH 0.35f
#define IOU_TH   0.8f

typedef __attribute__((ext_vector_type(16))) _Float16 v16h;
typedef __attribute__((ext_vector_type(8)))  float    v8f;
typedef __attribute__((ext_vector_type(4)))  unsigned u32x4;
typedef __attribute__((ext_vector_type(4)))  int      i32x4;
typedef __attribute__((ext_vector_type(8)))  int      i32x8;

#if defined(__has_builtin)
#if __has_builtin(__builtin_amdgcn_tensor_load_to_lds) && \
    __has_builtin(__builtin_amdgcn_s_wait_tensorcnt)
#define HAVE_TDM 1
#endif
#endif

// -------------------- workspace layout (in floats) -------------------------
#define OFF_XMIN   (0 * PN)
#define OFF_YMIN   (1 * PN)
#define OFF_XMAX   (2 * PN)
#define OFF_YMAX   (3 * PN)
#define OFF_AREA   (4 * PN)
#define OFF_VALID  (5 * PN)
#define OFF_CONFM  (6 * PN)
#define OFF_FIRST  (8 * PN)           // int32: colfirst = col_any ? first_i : -1
#define OFF_POSES  (9 * PN)           // [NPAIR*NBOX, 12]
#define OFF_SCORES (9 * PN + 12 * PN) // [BATCH, CN] mutable top-k scratch

__device__ __forceinline__ _Float16 h01(bool b) {
    return b ? (_Float16)1.0f : (_Float16)0.0f;   // branchless v_cndmask_b16
}

// ---------------------------------------------------------------------------
// Kernel 1: per (pair, box): enclosing 2D box, area, validity, masked conf,
// and the per-batch masked score scratch used by top-k.
// ---------------------------------------------------------------------------
__global__ void fd_prep_kernel(const float* __restrict__ boxes3D,
                               const float* __restrict__ classification,
                               const float* __restrict__ confidence,
                               float* __restrict__ ws) {
    int tid = blockIdx.x * blockDim.x + threadIdx.x;
    if (tid >= PN) return;
    int p = tid / NBOX;          // pair = b*NCLS + c
    int n = tid % NBOX;
    int b = p / NCLS;
    int c = p % NCLS;

    const float* bx = boxes3D + (((size_t)(b * NBOX + n) * NCLS + c) * 16);
    __builtin_prefetch((const void*)bx, 0, 3);   // global_prefetch_b8
    float xmn = bx[0], xmx = bx[0], ymn = bx[1], ymx = bx[1];
#pragma unroll
    for (int k = 1; k < 8; ++k) {
        float x = bx[2 * k], y = bx[2 * k + 1];
        xmn = fminf(xmn, x); xmx = fmaxf(xmx, x);
        ymn = fminf(ymn, y); ymx = fmaxf(ymx, y);
    }
    float score = classification[(size_t)(b * NBOX + n) * NCLS + c];
    float conf  = confidence[(size_t)(b * NBOX + n) * NCLS + c];
    bool  valid = score > SCORE_TH;

    ws[OFF_XMIN  + tid] = xmn;
    ws[OFF_YMIN  + tid] = ymn;
    ws[OFF_XMAX  + tid] = xmx;
    ws[OFF_YMAX  + tid] = ymx;
    ws[OFF_AREA  + tid] = (xmx - xmn + 1.0f) * (ymx - ymn + 1.0f);
    ws[OFF_VALID + tid] = valid ? 1.0f : 0.0f;
    ws[OFF_CONFM + tid] = valid ? conf : 0.0f;
    // scores_flat layout per batch: [C, N]
    ws[OFF_SCORES + (size_t)b * CN + c * NBOX + n] = valid ? score : -1.0f;
}

// ---------------------------------------------------------------------------
// Kernel 2: per column j, first row i with IoU(i,j) > 0.8 (validity-masked).
// One (pair, half) per 1024-thread block; the pair's 6 box arrays (48KB) are
// staged into LDS once and the O(N^2) scan runs entirely out of LDS.
// Writes colfirst[j] = hit ? i : -1  (fused indicator for kernel 3).
// ---------------------------------------------------------------------------
__global__ void __launch_bounds__(1024)
fd_firsthit_kernel(float* __restrict__ ws) {
    __shared__ float s_xmin[NBOX], s_ymin[NBOX], s_xmax[NBOX], s_ymax[NBOX];
    __shared__ float s_area[NBOX], s_val[NBOX];

    int p    = blockIdx.x >> 1;                       // pair
    int j    = ((blockIdx.x & 1) << 10) + threadIdx.x; // column handled
    int base = p * NBOX;

    for (int idx = threadIdx.x; idx < NBOX; idx += 1024) {
        s_xmin[idx] = ws[OFF_XMIN + base + idx];
        s_ymin[idx] = ws[OFF_YMIN + base + idx];
        s_xmax[idx] = ws[OFF_XMAX + base + idx];
        s_ymax[idx] = ws[OFF_YMAX + base + idx];
        s_area[idx] = ws[OFF_AREA + base + idx];
        s_val[idx]  = ws[OFF_VALID + base + idx];
    }
    __syncthreads();

    int* colfirst_g = (int*)(ws + OFF_FIRST);
    int  result = -1;
    if (s_val[j] != 0.0f) {
        float xminj = s_xmin[j], yminj = s_ymin[j];
        float xmaxj = s_xmax[j], ymaxj = s_ymax[j];
        float areaj = s_area[j];
        for (int i = 0; i <= j; ++i) {     // diagonal guarantees hit at i==j
            if (s_val[i] == 0.0f) continue;
            float x1 = fmaxf(xminj, s_xmin[i]);
            float y1 = fmaxf(yminj, s_ymin[i]);
            float x2 = fminf(xmaxj, s_xmax[i]);
            float y2 = fminf(ymaxj, s_ymax[i]);
            float wid = x2 - x1 + 1.0f;
            float hei = y2 - y1 + 1.0f;
            if (wid <= 0.0f || hei <= 0.0f) continue;
            float inter = wid * hei;
            float uni = areaj + s_area[i] - inter;
            float ov = (uni == 0.0f) ? 0.0f : (inter / uni);
            if (ov > IOU_TH) { result = i; break; }
        }
    }
    colfirst_g[base + j] = result;
}

// ---------------------------------------------------------------------------
// Kernel 3: one wave32 per 16-row strip of one (batch, class) pair.
//   (a) TDM-DMA colfirst[] into LDS (tensor_load_to_lds) while the wave
//       stages conf[] itself; s_wait_tensorcnt 0 before use.
//   (b) row_cnt = ind2 @ ones via v_wmma_f32_16x16x32_f16 (64 K-chunks,
//       branchless fragments from ds_load_b128 int4 reads).
//   (c) per-row stable 3-smallest scan of bc = ind2 * conf, pose averaging.
// ---------------------------------------------------------------------------
__global__ void __launch_bounds__(32)
fd_rowproc_kernel(const float* __restrict__ posesIn, float* __restrict__ ws) {
    __shared__ int   s_colfirst[NBOX];
    __shared__ float s_conf[NBOX];
    __shared__ float s_rowcnt[16];

    const int strips = NBOX / 16;            // 128
    int p       = blockIdx.x / strips;       // pair = b*NCLS + c
    int rowBase = (blockIdx.x % strips) * 16;
    int b = p / NCLS;
    int c = p % NCLS;
    int base = p * NBOX;
    int lane = threadIdx.x;                  // 0..31 (wave32)

    const int*   colfirst_g = (const int*)(ws + OFF_FIRST) + base;
    const float* confm_g    = ws + OFF_CONFM + base;

#ifdef HAVE_TDM
    {
        // D# group 0: count=1 | lds_addr | global_addr[56:0] | type=2
        unsigned lds_off = (unsigned)(size_t)
            (__attribute__((address_space(3))) int*)s_colfirst;
        unsigned long long ga = (unsigned long long)(const void*)colfirst_g;
        u32x4 g0;
        g0[0] = 1u;                                        // count=1 (valid)
        g0[1] = lds_off;                                   // lds_addr
        g0[2] = (unsigned)(ga & 0xFFFFFFFFu);              // global_addr lo
        g0[3] = (unsigned)((ga >> 32) & 0x01FFFFFFu) | (2u << 30); // hi|type=2
        // D# group 1: data_size=4B; tensor 2048x1; tile 2048x1; stride 2048
        i32x8 g1;
        g1[0] = 0x00020000;                  // wg_mask=0, data_size=2 (4B)
        g1[1] = (int)((NBOX & 0xFFFF) << 16);// tensor_dim0[15:0] @ bits 63:48
        g1[2] = (int)(1u << 16);             // dim0[31:16]=0 | tensor_dim1=1
        g1[3] = (int)((unsigned)NBOX << 16); // dim1 hi=0 | tile_dim0=2048
        g1[4] = 1;                           // tile_dim1=1, tile_dim2=0
        g1[5] = NBOX;                        // tensor_dim0_stride lo32
        g1[6] = 0;                           // stride hi | dim1_stride lo
        g1[7] = 0;
        i32x4 z4 = {0, 0, 0, 0};
#if __clang_major__ >= 23
        i32x8 z8 = {0, 0, 0, 0, 0, 0, 0, 0};
        __builtin_amdgcn_tensor_load_to_lds(g0, g1, z4, z4, z8, 0);
#else
        __builtin_amdgcn_tensor_load_to_lds(g0, g1, z4, z4, 0);
#endif
    }
    for (int idx = lane; idx < NBOX; idx += 32)   // overlap with TDM DMA
        s_conf[idx] = confm_g[idx];
    __builtin_amdgcn_s_wait_tensorcnt(0);
#else
    for (int idx = lane; idx < NBOX; idx += 32) {
        s_colfirst[idx] = colfirst_g[idx];
        s_conf[idx]     = confm_g[idx];
    }
#endif
    __syncthreads();

    // ---- WMMA row-count reduction: D = sum_k A(16x32 f16 indicator) * 1 ----
    int half = lane >> 4;                    // K half selector per A layout
    int irow = rowBase + (lane & 15);        // A-matrix row M = lane&15

    v16h ones;
#pragma unroll
    for (int t = 0; t < 16; ++t) ones[t] = (_Float16)1.0f;

    const int4* cf4 = (const int4*)s_colfirst;   // 16B-aligned ds_load_b128
    v8f acc = {};
#pragma unroll 4
    for (int chunk = 0; chunk < NBOX / 32; ++chunk) {
        // lane's 16 K-values = two contiguous 8-int runs:
        //   j = chunk*32 + half*8 + {0..7}  and  chunk*32 + 16 + half*8 + {0..7}
        int q = chunk * 8 + half * 2;
        int4 r0 = cf4[q];
        int4 r1 = cf4[q + 1];
        int4 r2 = cf4[q + 4];
        int4 r3 = cf4[q + 5];
        v16h a;
        a[0]  = h01(r0.x == irow); a[1]  = h01(r0.y == irow);
        a[2]  = h01(r0.z == irow); a[3]  = h01(r0.w == irow);
        a[4]  = h01(r1.x == irow); a[5]  = h01(r1.y == irow);
        a[6]  = h01(r1.z == irow); a[7]  = h01(r1.w == irow);
        a[8]  = h01(r2.x == irow); a[9]  = h01(r2.y == irow);
        a[10] = h01(r2.z == irow); a[11] = h01(r2.w == irow);
        a[12] = h01(r3.x == irow); a[13] = h01(r3.y == irow);
        a[14] = h01(r3.z == irow); a[15] = h01(r3.w == irow);
        acc = __builtin_amdgcn_wmma_f32_16x16x32_f16(
            false, a, false, ones, (short)0, acc, false, false);
    }
    // C/D layout: VGPR r -> M=r (lanes 0-15) / M=8+r (lanes 16-31).
    if (lane == 0) {
#pragma unroll
        for (int r = 0; r < 8; ++r) s_rowcnt[r] = acc[r];
    }
    if (lane == 16) {
#pragma unroll
        for (int r = 0; r < 8; ++r) s_rowcnt[8 + r] = acc[r];
    }
    __syncthreads();

    // ---- per-row: 3 smallest of bc[i,:] (stable, index-ascending ties) ----
    if (lane < 16) {
        int i = rowBase + lane;
        float b0 = 3.0e38f, b1 = 3.0e38f, b2 = 3.0e38f;
        int   i0 = 0, i1 = 0, i2 = 0;
        for (int j = 0; j < NBOX; ++j) {
            float v = (s_colfirst[j] == i) ? s_conf[j] : 0.0f;  // branchless
            if (v < b0)      { b2 = b1; i2 = i1; b1 = b0; i1 = i0; b0 = v; i0 = j; }
            else if (v < b1) { b2 = b1; i2 = i1; b1 = v; i1 = j; }
            else if (v < b2) { b2 = v; i2 = j; }
        }
        float rc = s_rowcnt[lane];
        float* po = ws + OFF_POSES + (size_t)(base + i) * 12;
        if (rc < 2.5f) {
#pragma unroll
            for (int k = 0; k < 12; ++k) po[k] = -1.0f;
        } else {
            const float* p0 = posesIn + ((size_t)(b * NBOX + i0) * NCLS + c) * 12;
            const float* p1 = posesIn + ((size_t)(b * NBOX + i1) * NCLS + c) * 12;
            const float* p2 = posesIn + ((size_t)(b * NBOX + i2) * NCLS + c) * 12;
#pragma unroll
            for (int k = 0; k < 12; ++k)
                po[k] = (p0[k] + p1[k] + p2[k]) * (1.0f / 3.0f);
        }
    }
}

// ---------------------------------------------------------------------------
// Kernel 4: per-batch top-300 (iterative argmax, index-ascending tie-break,
// matching lax.top_k) and final output assembly.
// ---------------------------------------------------------------------------
__global__ void __launch_bounds__(256)
fd_topk_kernel(float* __restrict__ ws, float* __restrict__ out) {
    __shared__ float s_val[256];
    __shared__ int   s_idx[256];

    int bth = blockIdx.x;                    // batch
    int t   = threadIdx.x;
    float* sc = ws + OFF_SCORES + (size_t)bth * CN;
    __builtin_prefetch((const void*)sc, 0, 3);

    float* scores_out  = out;                             // [B,300]
    int*   labels_out  = (int*)out + BATCH * MAXDET;      // [B,300]
    float* poses_out   = out + 2 * BATCH * MAXDET;        // [B,300,12]
    int*   indices_out = (int*)out + 2 * BATCH * MAXDET + BATCH * MAXDET * 12;

    for (int s = 0; s < MAXDET; ++s) {
        float best = -3.0e38f; int bi = CN;
        for (int idx = t; idx < CN; idx += 256) {
            float v = sc[idx];
            if (v > best || (v == best && idx < bi)) { best = v; bi = idx; }
        }
        s_val[t] = best; s_idx[t] = bi;
        __syncthreads();
        for (int stride = 128; stride > 0; stride >>= 1) {
            if (t < stride) {
                float ov = s_val[t + stride]; int oi = s_idx[t + stride];
                if (ov > s_val[t] || (ov == s_val[t] && oi < s_idx[t])) {
                    s_val[t] = ov; s_idx[t] = oi;
                }
            }
            __syncthreads();
        }
        if (t == 0) {
            float v = s_val[0]; int idx = s_idx[0];
            sc[idx] = -3.0e38f;                          // mark consumed
            bool mask = v > SCORE_TH;
            int cls = idx / NBOX, n = idx % NBOX;
            int o = bth * MAXDET + s;
            scores_out[o]  = mask ? v : -1.0f;
            labels_out[o]  = mask ? cls : -1;
            indices_out[o] = mask ? n : -1;
            const float* po = ws + OFF_POSES + (size_t)(bth * CN + idx) * 12;
#pragma unroll
            for (int k = 0; k < 12; ++k)
                poses_out[(size_t)o * 12 + k] = mask ? po[k] : -1.0f;
        }
        __syncthreads();
    }
}

// ---------------------------------------------------------------------------
extern "C" void kernel_launch(void* const* d_in, const int* in_sizes, int n_in,
                              void* d_out, int out_size, void* d_ws, size_t ws_size,
                              hipStream_t stream) {
    const float* boxes3D        = (const float*)d_in[0];   // [2,2048,8,16]
    const float* classification = (const float*)d_in[1];   // [2,2048,8]
    const float* poses          = (const float*)d_in[2];   // [2,2048,8,12]
    const float* confidence     = (const float*)d_in[3];   // [2,2048,8]
    float* ws  = (float*)d_ws;
    float* out = (float*)d_out;

    fd_prep_kernel<<<PN / 256, 256, 0, stream>>>(boxes3D, classification,
                                                 confidence, ws);
    fd_firsthit_kernel<<<NPAIR * 2, 1024, 0, stream>>>(ws);
    fd_rowproc_kernel<<<NPAIR * (NBOX / 16), 32, 0, stream>>>(poses, ws);
    fd_topk_kernel<<<BATCH, 256, 0, stream>>>(ws, out);
}
